// HeterGConvLayer_8993661518508
// MI455X (gfx1250) — matrix operand order repaired and checked by reading, not compiled, not run on hardware
//
#include <hip/hip_runtime.h>

typedef __attribute__((ext_vector_type(16))) _Float16 v16h;
typedef __attribute__((ext_vector_type(8)))  float    v8f;

#define FEAT 128

// -----------------------------------------------------------------------------
// Kernel 0: Wh[n][k] = (fp16) W[k][n]  — transposed fp16 copy of the weight so
// GEMM B-fragments (16 consecutive K for one column) become one b128-pair load.
// -----------------------------------------------------------------------------
__global__ __launch_bounds__(256)
void convert_W_t(const float* __restrict__ W, _Float16* __restrict__ Wh) {
    const int i = blockIdx.x * 256 + threadIdx.x;   // 0 .. 16383
    const int k = i >> 7;
    const int n = i & (FEAT - 1);
    Wh[(size_t)n * FEAT + k] = (_Float16)W[i];
}

// -----------------------------------------------------------------------------
// Kernel 1: support = feature @ W via v_wmma_f32_16x16x32_f16.
// 256 threads = 8 waves; each wave: 16 rows x 128 cols (8 tiles of 16x16).
// Tiles OUTER, K inner: A fragments (4 x v16h = 32 VGPRs) loaded once; only one
// 8-VGPR accumulator live at a time -> low register pressure, high occupancy.
// Store bounds check hoisted to one wave-uniform branch (781/782 blocks full).
// -----------------------------------------------------------------------------
__global__ __launch_bounds__(256)
void gemm_wmma_f16(const float* __restrict__ feat,
                   const _Float16* __restrict__ Wh,   // [N=128][K=128] fp16
                   float* __restrict__ support,
                   int nrows) {
    const int lane    = threadIdx.x & 31;
    const int half_id = lane >> 4;             // 0 or 1
    const int r       = lane & 15;             // 0..15
    const int m0      = blockIdx.x * 128 + (threadIdx.x >> 5) * 16;
    if (m0 >= nrows) return;                   // wave-uniform: WMMA always runs with EXEC all-1s
    const bool full = (m0 + 16) <= nrows;      // wave-uniform

    // ---- A fragments for all 4 K-chunks (contiguous K order, same as B) ----
    // element e<8  -> K = k0 + half_id*16 + e       (8 contiguous floats)
    // element e>=8 -> K = k0 + half_id*16 + e       (8 contiguous floats)
    int arow = m0 + r;
    if (arow >= nrows) arow = nrows - 1;       // clamp; garbage rows never stored
    const float* ap = feat + (size_t)arow * FEAT;
    v16h a[4];
#pragma unroll
    for (int kc = 0; kc < 4; ++kc) {
        const int k0 = kc * 32;
#pragma unroll
        for (int e = 0; e < 8; ++e)
            a[kc][e] = (_Float16)ap[k0 + half_id * 16 + e];
#pragma unroll
        for (int e = 0; e < 8; ++e)
            a[kc][8 + e] = (_Float16)ap[k0 + half_id * 16 + 8 + e];
    }

    // ---- 8 column tiles; per tile: 4x (b128-pair load + WMMA), then store ----
#pragma unroll
    for (int t = 0; t < 8; ++t) {
        const int n = t * 16 + r;                       // this lane's B column / store column
        const _Float16* bp = Wh + (size_t)n * FEAT;     // contiguous K for column n
        v8f acc = {};
#pragma unroll
        for (int kc = 0; kc < 4; ++kc) {
            // B element e -> K = kc*32 + half_id*16 + e : contiguous, 32B aligned
            const v16h b = *(const v16h*)(bp + kc * 32 + half_id * 16);
            acc = __builtin_amdgcn_wmma_f32_16x16x32_f16(
                /*neg_a=*/false, a[kc], /*neg_b=*/false, b,
                /*c_mod=*/(short)0, acc, /*reuse_a=*/false, /*reuse_b=*/false);
        }
        // C/D layout: VGPR j -> M = half_id*8 + j, N = r (within tile).
        // Base pointer + constant byte offsets (j*512 fits 24-bit IOFFSET).
        float* sp = support + (size_t)(m0 + half_id * 8) * FEAT + n;
        if (full) {
#pragma unroll
            for (int j = 0; j < 8; ++j)
                sp[j * FEAT] = acc[j];
        } else {
#pragma unroll
            for (int j = 0; j < 8; ++j)
                if (m0 + half_id * 8 + j < nrows) sp[j * FEAT] = acc[j];
        }
    }
}

// -----------------------------------------------------------------------------
// Kernel 2: out[n, f] = bias[f]   (base value for the segment sum)
// -----------------------------------------------------------------------------
__global__ __launch_bounds__(256)
void init_bias(float* __restrict__ out, const float* __restrict__ bias, int total) {
    const int i = blockIdx.x * blockDim.x + threadIdx.x;
    if (i < total) out[i] = bias[i & (FEAT - 1)];
}

// -----------------------------------------------------------------------------
// Kernel 3: out[dst] += w_e * support[src].  One wave per edge; each lane owns
// a float4 of the 128 features -> 4x global_atomic_add_f32 into L2-resident out
// (support and out both fit in the 192 MB L2).
// -----------------------------------------------------------------------------
__global__ __launch_bounds__(256)
void spmm_scatter(const long long* __restrict__ ei,   // [2, E] int64: row0=dst, row1=src
                  const float* __restrict__ ew,
                  const float* __restrict__ support,
                  float* __restrict__ out,
                  long long E) {
    const int lane = threadIdx.x & 31;
    const long long e = (long long)blockIdx.x * 8 + (threadIdx.x >> 5);
    if (e >= E) return;

    const long long d = ei[e];       // dst
    const long long s = ei[E + e];   // src
    const float w = ew[e];

    const float4 v = *(const float4*)(support + s * FEAT + lane * 4);
    float* op = out + d * FEAT + lane * 4;
    atomicAdd(op + 0, w * v.x);
    atomicAdd(op + 1, w * v.y);
    atomicAdd(op + 2, w * v.z);
    atomicAdd(op + 3, w * v.w);
}

// -----------------------------------------------------------------------------
// Launch. Inputs: 0=feature [N*128] f32, 1=edge_index [2*E] i64, 2=edge_weight
// [E] f32, 3=weight [128*128] f32, 4=bias [128] f32, 5=num_modal (==3 -> SGConv
// path always taken).
// -----------------------------------------------------------------------------
extern "C" void kernel_launch(void* const* d_in, const int* in_sizes, int n_in,
                              void* d_out, int out_size, void* d_ws, size_t ws_size,
                              hipStream_t stream) {
    const float*     feature = (const float*)d_in[0];
    const long long* ei      = (const long long*)d_in[1];
    const float*     ew      = (const float*)d_in[2];
    const float*     W       = (const float*)d_in[3];
    const float*     bias    = (const float*)d_in[4];

    const int nrows   = in_sizes[0] / FEAT;       // 100000
    const long long E = (long long)in_sizes[2];   // 1600000

    float*    support = (float*)d_ws;                                // N*128 f32
    _Float16* Wh      = (_Float16*)(support + (size_t)nrows * FEAT); // 32 KB, 32B-aligned

    float* out = (float*)d_out;

    // 0) Wh = transpose(W) as fp16
    convert_W_t<<<dim3(64), 256, 0, stream>>>(W, Wh);

    // 1) support = feature @ W  (WMMA)
    gemm_wmma_f16<<<dim3((nrows + 127) / 128), 256, 0, stream>>>(feature, Wh, support, nrows);

    // 2) out = broadcast(bias)
    init_bias<<<dim3((out_size + 255) / 256), 256, 0, stream>>>(out, bias, out_size);

    // 3) out[dst] += w * support[src]
    spmm_scatter<<<dim3((unsigned)((E + 7) / 8)), 256, 0, stream>>>(ei, ew, support, out, E);
}
